// Set2TreeEdge_83013127897108
// MI455X (gfx1250) — compile-verified, matrix operand order checked by reading, not run on hardware
//
#include <hip/hip_runtime.h>
#include <hip/hip_bf16.h>

// ---------------------------------------------------------------------------
// Types
// ---------------------------------------------------------------------------
typedef __bf16 bf16;
typedef __attribute__((ext_vector_type(16))) __bf16 v16bf;
typedef __attribute__((ext_vector_type(8)))  __bf16 v8bf;
typedef __attribute__((ext_vector_type(8)))  float  v8f;

#define N_NODES   8192
#define N_EDGES   131072
#define DIM       256
#define EPS_BN    1e-5f

enum { ACT_NONE = 0, ACT_ELU = 1, ACT_RELU = 2 };

// gfx1250 async copy global->LDS (ASYNCcnt-tracked, no VGPR staging).
// lds32 = wave-relative LDS byte address (low 32 bits of generic pointer).
__device__ __forceinline__ void async_load_b128(unsigned lds32,
                                                const void* gptr) {
  asm volatile("global_load_async_to_lds_b128 %0, %1, off"
               :: "v"(lds32), "v"(gptr)
               : "memory");
}
__device__ __forceinline__ void wait_async0() {
  asm volatile("s_wait_asynccnt 0x0" ::: "memory");
}

// ---------------------------------------------------------------------------
// Fused GEMM:  Out[M,Nout] = act(A[M,K](bf16) * W[K,Nout](f32->bf16) + bias)
// BM=128 (8 waves x 16 rows), BN=64 (4 WMMA col subtiles), BK=32.
// Double-buffered LDS; A tiles arrive via GLOBAL_LOAD_ASYNC_TO_LDS_B128 and
// overlap the WMMAs; one barrier per K-step. All fragments are hoisted before
// the 4 WMMAs so they issue back-to-back after a single s_wait_dscnt.
// Optionally accumulates per-column sum / sum-of-squares for BatchNorm.
// ---------------------------------------------------------------------------
template <int ACT, bool STATS>
__global__ __launch_bounds__(256) void gemm_bias_act(
    const bf16* __restrict__ A, const float* __restrict__ W,
    const float* __restrict__ bias, bf16* __restrict__ Out,
    int M, int K, int Nout, float* __restrict__ gsum, float* __restrict__ gsq) {
  __shared__ alignas(16) bf16 As[2][128 * 32];  // A tiles, row-major [128][32]
  __shared__ alignas(16) bf16 Wt[2][64 * 32];   // W tiles transposed [n][k]
  __shared__ float csum[64];
  __shared__ float csq[64];

  const int tid    = threadIdx.x;
  const int wave   = tid >> 5;
  const int lane   = tid & 31;
  const int l15    = lane & 15;
  const int kh     = lane >> 4;          // which K-half this lane owns
  const int rowBlk = blockIdx.x * 128;
  const int colBlk = blockIdx.y * 64;

  if (STATS && tid < 64) { csum[tid] = 0.f; csq[tid] = 0.f; }

  // per-thread tile-fill coordinates
  const int ca_r0 = (tid * 2) >> 2;            // A chunk 0 row
  const int ca_c0 = ((tid * 2) & 3) * 8;       // A chunk 0 col
  const int ca_r1 = (tid * 2 + 1) >> 2;        // A chunk 1 row
  const int ca_c1 = ((tid * 2 + 1) & 3) * 8;   // A chunk 1 col
  const int wk    = (tid * 8) >> 6;            // W tile k-row (0..31)
  const int wn    = (tid * 8) & 63;            // W tile n0 (multiple of 8)

  // ---- tile fill helpers --------------------------------------------------
  auto issueA = [&](int k0, int buf) {
    const bf16* g0 = A + (size_t)(rowBlk + ca_r0) * K + (k0 + ca_c0);
    const bf16* g1 = A + (size_t)(rowBlk + ca_r1) * K + (k0 + ca_c1);
    async_load_b128((unsigned)(uintptr_t)&As[buf][ca_r0 * 32 + ca_c0], g0);
    async_load_b128((unsigned)(uintptr_t)&As[buf][ca_r1 * 32 + ca_c1], g1);
  };
  auto stageW = [&](int k0, int buf) {
    const float* ws = W + (size_t)(k0 + wk) * Nout + colBlk + wn;
    float4 f0 = *(const float4*)ws;
    float4 f1 = *(const float4*)(ws + 4);
    bf16 tmp[8] = {(bf16)f0.x, (bf16)f0.y, (bf16)f0.z, (bf16)f0.w,
                   (bf16)f1.x, (bf16)f1.y, (bf16)f1.z, (bf16)f1.w};
#pragma unroll
    for (int j = 0; j < 8; ++j) Wt[buf][(wn + j) * 32 + wk] = tmp[j];
  };

  v8f acc[4];
#pragma unroll
  for (int i = 0; i < 4; ++i) {
#pragma unroll
    for (int r = 0; r < 8; ++r) acc[i][r] = 0.f;
  }

  const int nK = K >> 5;  // K / 32

  // ---- prologue: fill tile 0 ---------------------------------------------
  issueA(0, 0);
  stageW(0, 0);
  wait_async0();
  __syncthreads();

  for (int kt = 0; kt < nK; ++kt) {
    const int cur = kt & 1;
    const bool hasNext = (kt + 1) < nK;
    if (hasNext) {             // overlap next-tile copy with current WMMAs
      issueA((kt + 1) * 32, cur ^ 1);
      stageW((kt + 1) * 32, cur ^ 1);
    }

    // ---- hoist all fragments, then 4 back-to-back WMMAs ------------------
    const int m = wave * 16 + l15;
    v8bf alo = *(const v8bf*)&As[cur][m * 32 + kh * 8];
    v8bf ahi = *(const v8bf*)&As[cur][m * 32 + 16 + kh * 8];
    v16bf afrag = __builtin_shufflevector(alo, ahi, 0, 1, 2, 3, 4, 5, 6, 7, 8,
                                          9, 10, 11, 12, 13, 14, 15);
    v16bf bfrag[4];
#pragma unroll
    for (int ct = 0; ct < 4; ++ct) {
      int n = ct * 16 + l15;
      v8bf blo = *(const v8bf*)&Wt[cur][n * 32 + kh * 8];
      v8bf bhi = *(const v8bf*)&Wt[cur][n * 32 + 16 + kh * 8];
      bfrag[ct] = __builtin_shufflevector(blo, bhi, 0, 1, 2, 3, 4, 5, 6, 7, 8,
                                          9, 10, 11, 12, 13, 14, 15);
    }
#pragma unroll
    for (int ct = 0; ct < 4; ++ct) {
      acc[ct] = __builtin_amdgcn_wmma_f32_16x16x32_bf16(
          false, afrag, false, bfrag[ct], (short)0, acc[ct], false, false);
    }

    if (hasNext) wait_async0();  // next A tile resident in LDS
    __syncthreads();             // everyone done reading cur; next W visible
  }

  // ---- epilogue: bias + activation + store + column stats ----------------
#pragma unroll
  for (int ct = 0; ct < 4; ++ct) {
    int n = colBlk + ct * 16 + l15;
    float bb = bias[n];
    float s = 0.f, q = 0.f;
#pragma unroll
    for (int r = 0; r < 8; ++r) {
      float v = acc[ct][r] + bb;
      if (ACT == ACT_ELU) v = v > 0.f ? v : expm1f(v);
      if (ACT == ACT_RELU) v = fmaxf(v, 0.f);
      int mrow = rowBlk + wave * 16 + r + (kh ? 8 : 0);
      Out[(size_t)mrow * Nout + n] = (bf16)v;
      if (STATS) { s += v; q += v * v; }
    }
    if (STATS) {
      s += __shfl_xor(s, 16, 32);
      q += __shfl_xor(q, 16, 32);
      if (kh == 0) {
        atomicAdd(&csum[ct * 16 + l15], s);
        atomicAdd(&csq[ct * 16 + l15], q);
      }
    }
  }
  if (STATS) {
    __syncthreads();
    if (tid < 64) atomicAdd(&gsum[colBlk + tid], csum[tid]);
    else if (tid < 128) atomicAdd(&gsq[colBlk + tid - 64], csq[tid - 64]);
  }
}

// ---------------------------------------------------------------------------
// Small helper kernels
// ---------------------------------------------------------------------------
__global__ void zero_f32(float* __restrict__ p, size_t n) {
  for (size_t i = (size_t)blockIdx.x * blockDim.x + threadIdx.x; i < n;
       i += (size_t)gridDim.x * blockDim.x)
    p[i] = 0.f;
}

// x[N,16] f32 -> xpad[N,32] bf16 (zero-padded K)
__global__ __launch_bounds__(256) void pad_x(const float* __restrict__ x,
                                             bf16* __restrict__ xp, int N) {
  int t = blockIdx.x * 256 + threadIdx.x;
  int r = t >> 5, c = t & 31;
  if (r >= N) return;
  xp[t] = (bf16)((c < 16) ? x[r * 16 + c] : 0.f);
}

// BN: scale = gamma*rsqrt(var+eps), shift = beta - mu*scale   (256 cols)
__global__ void bn_finalize(const float* __restrict__ sum,
                            const float* __restrict__ sq,
                            const float* __restrict__ gamma,
                            const float* __restrict__ beta, float invM,
                            float* __restrict__ scale,
                            float* __restrict__ shift) {
  int n = threadIdx.x;
  float mu  = sum[n] * invM;
  float var = fmaxf(sq[n] * invM - mu * mu, 0.f);
  float sc  = gamma[n] * rsqrtf(var + EPS_BN);
  scale[n] = sc;
  shift[n] = beta[n] - mu * sc;
}

// y = x*scale[col] + shift[col], in place, C==256
__global__ __launch_bounds__(256) void bn_apply(bf16* __restrict__ x,
                                                const float* __restrict__ scale,
                                                const float* __restrict__ shift,
                                                size_t total) {
  size_t i = ((size_t)blockIdx.x * blockDim.x + threadIdx.x) * 8;
  if (i >= total) return;
  v8bf v = *(v8bf*)(x + i);
  int c = (int)(i & 255);
#pragma unroll
  for (int j = 0; j < 8; ++j)
    v[j] = (bf16)((float)v[j] * scale[c + j] + shift[c + j]);
  *(v8bf*)(x + i) = v;
}

// out[e] = [h[src[e]], h[dst[e]]]   (node-gather concat, C=256)
__global__ __launch_bounds__(256) void concat_gather(
    const bf16* __restrict__ h, const int* __restrict__ src,
    const int* __restrict__ dst, bf16* __restrict__ out, int E, int C) {
  int t = blockIdx.x * 256 + threadIdx.x;
  int cpe = (2 * C) / 8;
  int e = t / cpe, c = (t % cpe) * 8;
  if (e >= E) return;
  const bf16* sp = (c < C) ? (h + (size_t)src[e] * C + c)
                           : (h + (size_t)dst[e] * C + (c - C));
  *(v8bf*)(out + (size_t)e * 2 * C + c) = *(const v8bf*)sp;
}

// out[e] = [a[e], b[e]]   (edge-edge concat)
__global__ __launch_bounds__(256) void concat_edges(
    const bf16* __restrict__ a, const bf16* __restrict__ b,
    bf16* __restrict__ out, int E, int C) {
  int t = blockIdx.x * 256 + threadIdx.x;
  int cpe = (2 * C) / 8;
  int e = t / cpe, c = (t % cpe) * 8;
  if (e >= E) return;
  const bf16* sp = (c < C) ? (a + (size_t)e * C + c)
                           : (b + (size_t)e * C + (c - C));
  *(v8bf*)(out + (size_t)e * 2 * C + c) = *(const v8bf*)sp;
}

// segment sum (f32 atomics) + count
__global__ __launch_bounds__(256) void seg_accum(
    const bf16* __restrict__ e, const int* __restrict__ idx,
    float* __restrict__ acc, float* __restrict__ cnt, int E, int C) {
  int t = blockIdx.x * 256 + threadIdx.x;
  int cpe = C / 8;
  int ed = t / cpe, c = (t % cpe) * 8;
  if (ed >= E) return;
  int node = idx[ed];
  v8bf v = *(const v8bf*)(e + (size_t)ed * C + c);
  float* a = acc + (size_t)node * C + c;
#pragma unroll
  for (int j = 0; j < 8; ++j) atomicAdd(a + j, (float)v[j]);
  if (c == 0) atomicAdd(&cnt[node], 1.f);
}

__global__ __launch_bounds__(256) void seg_div(const float* __restrict__ acc,
                                               const float* __restrict__ cnt,
                                               bf16* __restrict__ out, int N,
                                               int C) {
  int t = blockIdx.x * 256 + threadIdx.x;
  int cpe = C / 8;
  int node = t / cpe, c = (t % cpe) * 8;
  if (node >= N) return;
  float inv = 1.f / fmaxf(cnt[node], 1.f);
  const float* a = acc + (size_t)node * C + c;
  v8bf o;
#pragma unroll
  for (int j = 0; j < 8; ++j) o[j] = (bf16)(a[j] * inv);
  *(v8bf*)(out + (size_t)node * C + c) = o;
}

// out[e] = h[dst[e]] - h[src[e]]
__global__ __launch_bounds__(256) void edge_diff(
    const bf16* __restrict__ h, const int* __restrict__ src,
    const int* __restrict__ dst, bf16* __restrict__ out, int E, int C) {
  int t = blockIdx.x * 256 + threadIdx.x;
  int cpe = C / 8;
  int e = t / cpe, c = (t % cpe) * 8;
  if (e >= E) return;
  v8bf vd = *(const v8bf*)(h + (size_t)dst[e] * C + c);
  v8bf vs = *(const v8bf*)(h + (size_t)src[e] * C + c);
  v8bf o;
#pragma unroll
  for (int j = 0; j < 8; ++j) o[j] = (bf16)((float)vd[j] - (float)vs[j]);
  *(v8bf*)(out + (size_t)e * C + c) = o;
}

// out = relu(a+b), with column stats; one thread per column, 8 rows per block
__global__ __launch_bounds__(256) void add_relu_stats(
    const bf16* __restrict__ a, const bf16* __restrict__ b,
    bf16* __restrict__ out, float* __restrict__ gsum, float* __restrict__ gsq,
    int M) {
  int rowBase = blockIdx.x * 8;
  int c = threadIdx.x;
  float s = 0.f, q = 0.f;
#pragma unroll
  for (int r = 0; r < 8; ++r) {
    size_t i = (size_t)(rowBase + r) * 256 + c;
    float v = fmaxf((float)a[i] + (float)b[i], 0.f);
    out[i] = (bf16)v;
    s += v;
    q += v * v;
  }
  atomicAdd(&gsum[c], s);
  atomicAdd(&gsq[c], q);
}

// fc_out: [E,256] x [256,8] + b -> [E,8] f32
__global__ __launch_bounds__(256) void fc_out_kernel(
    const bf16* __restrict__ e, const float* __restrict__ w,
    const float* __restrict__ b, float* __restrict__ out, int E) {
  int idx = blockIdx.x * 256 + threadIdx.x;
  int ed = idx >> 3, cls = idx & 7;
  if (ed >= E) return;
  float acc = b[cls];
  const bf16* row = e + (size_t)ed * 256;
  for (int k = 0; k < 256; ++k) acc += (float)row[k] * w[k * 8 + cls];
  out[idx] = acc;
}

// ---------------------------------------------------------------------------
// Host-side orchestration
// ---------------------------------------------------------------------------
struct Mlp { const float *w1, *b1, *w2, *b2, *g, *be; };

static Mlp take_mlp(void* const* d, int& i) {
  Mlp m{(const float*)d[i],     (const float*)d[i + 1],
        (const float*)d[i + 2], (const float*)d[i + 3],
        (const float*)d[i + 4], (const float*)d[i + 5]};
  i += 6;
  return m;
}

struct Stats { float *sum, *sq, *scale, *shift; };

// One NRI _mlp: fc1+ELU -> fc2+ELU (+stats) -> bn finalize -> bn apply
static void run_mlp(hipStream_t s, const bf16* A, int M, int K, const Mlp& p,
                    bf16* H1, bf16* OUT, const Stats& st) {
  zero_f32<<<1, 256, 0, s>>>(st.sum, 512);  // sum(256) + sq(256) contiguous
  dim3 grid(M / 128, DIM / 64);
  gemm_bias_act<ACT_ELU, false><<<grid, 256, 0, s>>>(A, p.w1, p.b1, H1, M, K,
                                                     DIM, nullptr, nullptr);
  gemm_bias_act<ACT_ELU, true><<<grid, 256, 0, s>>>(H1, p.w2, p.b2, OUT, M,
                                                    DIM, DIM, st.sum, st.sq);
  bn_finalize<<<1, 256, 0, s>>>(st.sum, st.sq, p.g, p.be, 1.f / (float)M,
                                st.scale, st.shift);
  size_t total = (size_t)M * DIM;
  bn_apply<<<(unsigned)((total / 8 + 255) / 256), 256, 0, s>>>(OUT, st.scale,
                                                               st.shift, total);
}

extern "C" void kernel_launch(void* const* d_in, const int* in_sizes, int n_in,
                              void* d_out, int out_size, void* d_ws,
                              size_t ws_size, hipStream_t stream) {
  (void)in_sizes; (void)n_in; (void)out_size; (void)ws_size;
  const int N = N_NODES, E = N_EDGES;

  // ---- inputs (insertion order of setup_inputs) --------------------------
  int ii = 0;
  const float* x   = (const float*)d_in[ii++];
  const int*   src = (const int*)d_in[ii++];
  const int*   dst = (const int*)d_in[ii++];
  Mlp init0 = take_mlp(d_in, ii);
  Mlp init1 = take_mlp(d_in, ii);
  Mlp pre   = take_mlp(d_in, ii);
  Mlp bnode[2][2], bedge1[2], bedge2[2];
  for (int b = 0; b < 2; ++b) {
    bnode[b][0] = take_mlp(d_in, ii);
    bnode[b][1] = take_mlp(d_in, ii);
    bedge1[b]   = take_mlp(d_in, ii);
    bedge2[b]   = take_mlp(d_in, ii);
  }
  const float* ec_tw = (const float*)d_in[ii++];
  const float* ec_tb = (const float*)d_in[ii++];
  const float* ec_pw = (const float*)d_in[ii++];
  const float* ec_pb = (const float*)d_in[ii++];
  const float* ec_g  = (const float*)d_in[ii++];
  const float* ec_be = (const float*)d_in[ii++];
  Mlp fin0 = take_mlp(d_in, ii);
  Mlp fin1 = take_mlp(d_in, ii);
  const float* fcw = (const float*)d_in[ii++];
  const float* fcb = (const float*)d_in[ii++];

  // ---- workspace carve (256B aligned) ------------------------------------
  size_t off = 0;
  auto carve = [&](size_t bytes) -> void* {
    void* p = (char*)d_ws + off;
    off += (bytes + 255) & ~(size_t)255;
    return p;
  };
  bf16*  xpad  = (bf16*)carve((size_t)N * 32 * sizeof(bf16));
  bf16*  hA    = (bf16*)carve((size_t)N * DIM * sizeof(bf16));
  bf16*  hB    = (bf16*)carve((size_t)N * DIM * sizeof(bf16));
  float* nAcc  = (float*)carve((size_t)N * (DIM + 1) * sizeof(float));
  float* nCnt  = nAcc + (size_t)N * DIM;
  bf16*  eb0   = (bf16*)carve((size_t)E * 512 * sizeof(bf16));  // concat buf
  bf16*  eb1   = (bf16*)carve((size_t)E * DIM * sizeof(bf16));  // H1 temp
  bf16*  eb2   = (bf16*)carve((size_t)E * DIM * sizeof(bf16));  // e (current)
  bf16*  eb3   = (bf16*)carve((size_t)E * DIM * sizeof(bf16));  // global skip
  bf16*  eb4   = (bf16*)carve((size_t)E * DIM * sizeof(bf16));  // x_skip/diff
  bf16*  eb5   = (bf16*)carve((size_t)E * DIM * sizeof(bf16));  // temp
  float* ssum  = (float*)carve(512 * sizeof(float));
  float* ssq   = ssum + 256;
  float* scal  = (float*)carve(512 * sizeof(float));
  float* shif  = scal + 256;
  Stats st{ssum, ssq, scal, shif};

  const size_t eBytes = (size_t)E * DIM * sizeof(bf16);

  // ---- pipeline -----------------------------------------------------------
  pad_x<<<N * 32 / 256, 256, 0, stream>>>(x, xpad, N);
  run_mlp(stream, xpad, N, 32, init0, hB, hA, st);     // initial[0] (K padded)
  run_mlp(stream, hA, N, DIM, init1, hB, hA, st);      // initial[1]

  concat_gather<<<E * 64 / 256, 256, 0, stream>>>(hA, src, dst, eb0, E, DIM);
  run_mlp(stream, eb0, E, 2 * DIM, pre, eb1, eb2, st); // e = pre(concat)
  hipMemcpyAsync(eb3, eb2, eBytes, hipMemcpyDeviceToDevice, stream); // skip

  for (int b = 0; b < 2; ++b) {
    hipMemcpyAsync(eb4, eb2, eBytes, hipMemcpyDeviceToDevice, stream); // x_skip
    zero_f32<<<2048, 256, 0, stream>>>(nAcc, (size_t)N * (DIM + 1));
    seg_accum<<<E * 32 / 256, 256, 0, stream>>>(eb2, dst, nAcc, nCnt, E, DIM);
    seg_div<<<N * 32 / 256, 256, 0, stream>>>(nAcc, nCnt, hA, N, DIM);
    run_mlp(stream, hA, N, DIM, bnode[b][0], hB, hA, st);
    run_mlp(stream, hA, N, DIM, bnode[b][1], hB, hA, st);
    concat_gather<<<E * 64 / 256, 256, 0, stream>>>(hA, src, dst, eb0, E, DIM);
    run_mlp(stream, eb0, E, 2 * DIM, bedge1[b], eb1, eb5, st);   // e2
    concat_edges<<<E * 64 / 256, 256, 0, stream>>>(eb5, eb4, eb0, E, DIM);
    run_mlp(stream, eb0, E, 2 * DIM, bedge2[b], eb1, eb2, st);   // new e
  }

  // EdgeConv
  zero_f32<<<2048, 256, 0, stream>>>(nAcc, (size_t)N * (DIM + 1));
  seg_accum<<<E * 32 / 256, 256, 0, stream>>>(eb2, dst, nAcc, nCnt, E, DIM);
  seg_div<<<N * 32 / 256, 256, 0, stream>>>(nAcc, nCnt, hA, N, DIM);
  edge_diff<<<E * 32 / 256, 256, 0, stream>>>(hA, src, dst, eb4, E, DIM);
  zero_f32<<<1, 256, 0, stream>>>(ssum, 512);
  {
    dim3 grid(E / 128, DIM / 64);
    gemm_bias_act<ACT_NONE, false><<<grid, 256, 0, stream>>>(
        eb4, ec_tw, ec_tb, eb1, E, DIM, DIM, nullptr, nullptr);
    gemm_bias_act<ACT_NONE, false><<<grid, 256, 0, stream>>>(
        eb2, ec_pw, ec_pb, eb5, E, DIM, DIM, nullptr, nullptr);
  }
  add_relu_stats<<<E / 8, 256, 0, stream>>>(eb1, eb5, eb2, ssum, ssq, E);
  bn_finalize<<<1, 256, 0, stream>>>(ssum, ssq, ec_g, ec_be, 1.f / (float)E,
                                     scal, shif);
  bn_apply<<<(unsigned)(((size_t)E * DIM / 8 + 255) / 256), 256, 0, stream>>>(
      eb2, scal, shif, (size_t)E * DIM);

  // final stack, applied twice with global skip
  for (int rep = 0; rep < 2; ++rep) {
    concat_edges<<<E * 64 / 256, 256, 0, stream>>>(eb2, eb3, eb0, E, DIM);
    run_mlp(stream, eb0, E, 2 * DIM, fin0, eb1, eb4, st);
    run_mlp(stream, eb4, E, DIM, fin1, eb1, eb2, st);
  }

  fc_out_kernel<<<E * 8 / 256, 256, 0, stream>>>(eb2, fcw, fcb, (float*)d_out,
                                                 E);
}
